// CenterHead_67190468379130
// MI455X (gfx1250) — matrix operand order repaired and checked by reading, not compiled, not run on hardware
//
#include <hip/hip_runtime.h>
#include <hip/hip_bf16.h>
#include <math.h>

// ---------------------------------------------------------------------------
// CenterHead for MI455X (gfx1250): bf16 WMMA implicit-GEMM conv pipeline with
// async global->LDS staging (ASYNCcnt) and double-buffered LDS.
// ---------------------------------------------------------------------------

typedef __attribute__((ext_vector_type(16))) __bf16          v16bf;
typedef __attribute__((ext_vector_type(8)))  float           v8f;
typedef __attribute__((ext_vector_type(8)))  unsigned short  ush8;

union Frag { v16bf v; ush8 h[2]; };

static __device__ __forceinline__ unsigned short f2bf_bits(float f) {
    unsigned int u = __float_as_uint(f);
    u += 0x7FFFu + ((u >> 16) & 1u);           // round-to-nearest-even
    return (unsigned short)(u >> 16);
}

// Async copy of 16 bytes global -> LDS (tracked by ASYNCcnt).
// lds_byte is the LDS byte address (single __shared__ alloc => base offset 0).
static __device__ __forceinline__ void async_ld16(unsigned lds_byte, const void* gaddr) {
    asm volatile("global_load_async_to_lds_b128 %0, %1, off"
                 :: "v"(lds_byte), "v"(gaddr) : "memory");
}
static __device__ __forceinline__ void wait_async0() {
    asm volatile("s_wait_asynccnt 0" ::: "memory");
}

#define HW_H   448
#define HW_W   448
#define HW_S   (448 * 448)   // 200704
#define CIN    256
#define COUT1  64
#define NB     4
#define TOPK   500

// ---------------------------------------------------------------------------
// Pack kernels: one-time f32 -> bf16 re-layouts so the WMMA kernels stage
// contiguous, conversion-free 16B chunks (async-LDS friendly).
// ---------------------------------------------------------------------------

// x [B][256][H][W] f32  ->  xp [B][cc=8][H][W][32] bf16 (LDS-transposed write)
__global__ __launch_bounds__(256)
void pack_x_kernel(const float* __restrict__ x, unsigned short* __restrict__ xp) {
    const int xt = blockIdx.x * 64;
    const int yy = blockIdx.y;
    const int bz = blockIdx.z;            // b*8 + cc
    const int b  = bz >> 3, cc = bz & 7;
    __shared__ unsigned short t[64 * 32];
    for (int e = threadIdx.x; e < 2048; e += 256) {
        int xi = e % 64;                  // coalesced along W
        int k  = e / 64;
        float v = x[(((size_t)b * CIN + cc * 32 + k) * HW_H + yy) * HW_W + xt + xi];
        t[xi * 32 + k] = f2bf_bits(v);
    }
    __syncthreads();
    {   // 256 chunks of 16B, coalesced
        int ck = threadIdx.x & 3;
        int xi = threadIdx.x >> 2;
        *(ush8*)&xp[((((size_t)b * 8 + cc) * HW_H + yy) * HW_W + (xt + xi)) * 32 + ck * 8] =
            *(const ush8*)&t[xi * 32 + ck * 8];
    }
}

// sc_w [64][256][3][3] f32 -> wp [cc=8][tap=9][cout=64][cin=32] bf16
__global__ __launch_bounds__(256)
void pack_w_kernel(const float* __restrict__ w, unsigned short* __restrict__ wp) {
    int i = blockIdx.x * 256 + threadIdx.x;
    if (i >= 8 * 9 * 64 * 32) return;
    int k = i % 32, r = i / 32;
    int co = r % 64; r /= 64;
    int t = r % 9, cc = r / 9;
    wp[i] = f2bf_bits(w[((size_t)co * CIN + cc * 32 + k) * 9 + t]);
}

// fused head weights -> wp2 [cc=2][tap=9][cout=16][cin=32] bf16 (pad 11..15)
__global__ __launch_bounds__(256)
void pack_w2_kernel(const float* __restrict__ hm_w, const float* __restrict__ ce_w,
                    const float* __restrict__ cz_w, const float* __restrict__ dm_w,
                    const float* __restrict__ rt_w, unsigned short* __restrict__ wp2) {
    int i = blockIdx.x * 256 + threadIdx.x;
    if (i >= 2 * 9 * 16 * 32) return;
    int k = i % 32, r = i / 32;
    int co = r % 16; r /= 16;
    int t = r % 9, cc = r / 9;
    int cin = cc * 32 + k;
    float v = 0.0f;
    if      (co < 3)  v = hm_w[((size_t)co       * 64 + cin) * 9 + t];
    else if (co < 5)  v = ce_w[((size_t)(co - 3) * 64 + cin) * 9 + t];
    else if (co < 6)  v = cz_w[((size_t)cin) * 9 + t];
    else if (co < 9)  v = dm_w[((size_t)(co - 6) * 64 + cin) * 9 + t];
    else if (co < 11) v = rt_w[((size_t)(co - 9) * 64 + cin) * 9 + t];
    wp2[i] = f2bf_bits(v);
}

// ---------------------------------------------------------------------------
// Kernel 1: shared 3x3 conv (256->64) + BN + ReLU.
// 256 threads = 8 waves; tile 64 couts x 32 pixels; double-buffered LDS fed by
// global_load_async_to_lds_b128; 72 v_wmma_f32_16x16x32_bf16 per wave with a
// register-pipelined tap loop. Output y in [B][H][W][64] bf16.
// ---------------------------------------------------------------------------
#define SWH 18432                 // halves per sW buffer (9*64*32)
#define SXH 3264                  // halves per sX buffer (3*34*32)
#define SXBASE_H (2 * SWH)        // sX region start (halves)
#define SWB(buf)  ((unsigned)((buf) * SWH * 2))              // byte offsets
#define SXB(buf)  ((unsigned)((SXBASE_H + (buf) * SXH) * 2))
#define XPLANE_B  ((size_t)HW_S * 32 * 2)                    // bytes per (b,cc) plane

__global__ __launch_bounds__(256)
void conv1_wmma_kernel(const unsigned short* __restrict__ xp,
                       const unsigned short* __restrict__ wp,
                       const float* __restrict__ cb,
                       const float* __restrict__ gamma,
                       const float* __restrict__ beta,
                       const float* __restrict__ mean,
                       const float* __restrict__ var,
                       unsigned short* __restrict__ y) {
    const int b  = blockIdx.z;
    const int yy = blockIdx.y;
    const int x0 = blockIdx.x * 32;
    const int tid  = threadIdx.x;
    const int lane = tid & 31;
    const int wv   = tid >> 5;
    const int mt   = wv >> 1;      // cout tile 0..3
    const int nt   = wv & 1;       // pixel tile 0..1

    __shared__ unsigned short smem[2 * SWH + 2 * SXH];   // sole LDS alloc (offset 0)

    const char* xbase = (const char*)xp + (size_t)b * 8 * XPLANE_B;
    const char* wbase = (const char*)wp;

    // Per-thread sX staging slots: 408 16B chunks -> up to 2 per thread.
    int      s_ok[2];
    unsigned s_loff[2];            // byte offset inside an sX buffer
    size_t   s_goff[2];            // byte offset inside one (b,cc) plane
    #pragma unroll
    for (int u = 0; u < 2; ++u) {
        int e = tid + u * 256;
        s_ok[u] = 0; s_loff[u] = 0; s_goff[u] = 0;
        if (e < 408) {
            int ck = e & 3, rc = e >> 2;
            int c = rc % 34, r = rc / 34;
            int sy = yy + r - 1, sx = x0 + c - 1;
            s_loff[u] = (unsigned)(((r * 34 + c) * 32 + ck * 8) * 2);
            if (sy >= 0 && sy < HW_H && sx >= 0 && sx < HW_W) {
                s_ok[u] = 1;
                s_goff[u] = ((size_t)(sy * HW_W + sx) * 32 + ck * 8) * 2;
            } else {
                // halo slots: zero both buffers once; async never writes them
                ush8 z = {0, 0, 0, 0, 0, 0, 0, 0};
                *(ush8*)((char*)smem + SXB(0) + s_loff[u]) = z;
                *(ush8*)((char*)smem + SXB(1) + s_loff[u]) = z;
            }
        }
    }

    // ---- stage chunk 0 into buffer 0 (async) ----
    #pragma unroll
    for (int i = 0; i < 9; ++i) {                 // 2304 = 9*256 weight chunks
        int e = tid + i * 256;
        async_ld16(SWB(0) + e * 16, wbase + (size_t)e * 16);
    }
    #pragma unroll
    for (int u = 0; u < 2; ++u)
        if (s_ok[u]) async_ld16(SXB(0) + s_loff[u], xbase + s_goff[u]);
    wait_async0();
    __syncthreads();

    v8f acc;
    #pragma unroll
    for (int i = 0; i < 8; ++i) acc[i] = 0.0f;

    const int Ka = (lane >= 16) ? 8  : 0;    // 16-bit A-matrix K base
    const int Kb = (lane >= 16) ? 8  : 0;    // 16-bit B-matrix K base
    const int M  = lane & 15;

    for (int cc = 0; cc < 8; ++cc) {
        const int cur = cc & 1, nxt = cur ^ 1;
        if (cc < 7) {   // prefetch next chunk into alternate buffers (async)
            #pragma unroll
            for (int i = 0; i < 9; ++i) {
                int e = tid + i * 256;
                async_ld16(SWB(nxt) + e * 16,
                           wbase + (size_t)(cc + 1) * (SWH * 2) + (size_t)e * 16);
            }
            #pragma unroll
            for (int u = 0; u < 2; ++u)
                if (s_ok[u])
                    async_ld16(SXB(nxt) + s_loff[u],
                               xbase + (size_t)(cc + 1) * XPLANE_B + s_goff[u]);
        }

        const unsigned short* sWp = smem + cur * SWH;
        const unsigned short* sXp = smem + SXBASE_H + cur * SXH;

        // register-pipelined tap loop: load t+1 fragments before WMMA(t)
        Frag a0, b0, a1, b1;
        {
            const ush8* ap = (const ush8*)&sWp[(mt * 16 + M) * 32 + Ka];
            a0.h[0] = ap[0]; a0.h[1] = ap[2];
            const ush8* bp = (const ush8*)&sXp[(nt * 16 + M) * 32 + Kb];
            b0.h[0] = bp[0]; b0.h[1] = bp[2];
        }
        #pragma unroll
        for (int t = 0; t < 9; ++t) {
            if (t < 8) {
                const int dy = (t + 1) / 3, dx = (t + 1) % 3;
                const ush8* ap = (const ush8*)&sWp[(((t + 1) * 64) + (mt * 16 + M)) * 32 + Ka];
                a1.h[0] = ap[0]; a1.h[1] = ap[2];
                const ush8* bp = (const ush8*)&sXp[((dy * 34) + (nt * 16 + M) + dx) * 32 + Kb];
                b1.h[0] = bp[0]; b1.h[1] = bp[2];
            }
            acc = __builtin_amdgcn_wmma_f32_16x16x32_bf16(
                      false, a0.v, false, b0.v, (short)0, acc, false, false);
            a0 = a1; b0 = b1;
        }

        if (cc < 7) wait_async0();
        __syncthreads();
    }

    // ---- epilogue: bias + BN(eval) + ReLU, pack 8 couts -> one b128 store ----
    const int p    = x0 + nt * 16 + (lane & 15);
    const int cbas = mt * 16 + ((lane >= 16) ? 8 : 0);
    ush8 outv;
    #pragma unroll
    for (int r = 0; r < 8; ++r) {
        const int ch = cbas + r;
        float s = gamma[ch] * rsqrtf(var[ch] + 1e-5f);
        float v = (acc[r] + cb[ch] - mean[ch]) * s + beta[ch];
        v = fmaxf(v, 0.0f);
        outv[r] = f2bf_bits(v);
    }
    *(ush8*)&y[(((size_t)b * HW_H + yy) * HW_W + p) * COUT1 + cbas] = outv;
}

// ---------------------------------------------------------------------------
// Kernel 2: all 5 heads fused as one 3x3 conv (64 -> 16, 11 real channels).
// 128 threads = 4 waves, tile 16 couts x 64 pixels, async LDS staging.
// ---------------------------------------------------------------------------
#define S2WH 4608                 // halves: 9*16*32
#define S2XH 6336                 // halves: 3*66*32
#define S2XB ((unsigned)(S2WH * 2))

__global__ __launch_bounds__(128)
void heads_wmma_kernel(const unsigned short* __restrict__ y,
                       const unsigned short* __restrict__ wp2,
                       const float* __restrict__ hm_b, const float* __restrict__ ce_b,
                       const float* __restrict__ cz_b, const float* __restrict__ dm_b,
                       const float* __restrict__ rt_b,
                       float* __restrict__ hm, float* __restrict__ reg) {
    const int b  = blockIdx.z;
    const int yy = blockIdx.y;
    const int x0 = blockIdx.x * 64;
    const int tid  = threadIdx.x;
    const int lane = tid & 31;
    const int nt   = tid >> 5;     // pixel tile 0..3

    __shared__ unsigned short smem[S2WH + S2XH];   // sole LDS alloc (offset 0)

    v8f acc;
    #pragma unroll
    for (int i = 0; i < 8; ++i) acc[i] = 0.0f;

    const int Ka = (lane >= 16) ? 8  : 0;
    const int Kb = (lane >= 16) ? 8  : 0;
    const int M  = lane & 15;

    // zero halo slots once (792 sX chunks; OOB ones persist as zeros)
    for (int e = tid; e < 792; e += 128) {
        int ck = e & 3, rc = e >> 2;
        int c = rc % 66, r = rc / 66;
        int sy = yy + r - 1, sx = x0 + c - 1;
        if (!(sy >= 0 && sy < HW_H && sx >= 0 && sx < HW_W)) {
            ush8 z = {0, 0, 0, 0, 0, 0, 0, 0};
            *(ush8*)&smem[S2WH + (r * 66 + c) * 32 + ck * 8] = z;
        }
    }
    __syncthreads();

    for (int cc = 0; cc < 2; ++cc) {
        // weights: 576 16B chunks
        for (int e = tid; e < 576; e += 128)
            async_ld16((unsigned)(e * 16), (const char*)wp2 + (size_t)cc * 9216 + e * 16);
        // activations: contiguous channel-last bf16 y
        for (int e = tid; e < 792; e += 128) {
            int ck = e & 3, rc = e >> 2;
            int c = rc % 66, r = rc / 66;
            int sy = yy + r - 1, sx = x0 + c - 1;
            if (sy >= 0 && sy < HW_H && sx >= 0 && sx < HW_W)
                async_ld16(S2XB + (unsigned)(((r * 66 + c) * 32 + ck * 8) * 2),
                           (const char*)y +
                           ((((size_t)b * HW_H + sy) * HW_W + sx) * COUT1
                            + cc * 32 + ck * 8) * 2);
        }
        wait_async0();
        __syncthreads();

        const unsigned short* sWp = smem;
        const unsigned short* sXp = smem + S2WH;
        Frag a0, b0, a1, b1;
        {
            const ush8* ap = (const ush8*)&sWp[M * 32 + Ka];
            a0.h[0] = ap[0]; a0.h[1] = ap[2];
            const ush8* bp = (const ush8*)&sXp[(nt * 16 + M) * 32 + Kb];
            b0.h[0] = bp[0]; b0.h[1] = bp[2];
        }
        #pragma unroll
        for (int t = 0; t < 9; ++t) {
            if (t < 8) {
                const int dy = (t + 1) / 3, dx = (t + 1) % 3;
                const ush8* ap = (const ush8*)&sWp[(((t + 1) * 16) + M) * 32 + Ka];
                a1.h[0] = ap[0]; a1.h[1] = ap[2];
                const ush8* bp = (const ush8*)&sXp[((dy * 66) + (nt * 16 + M) + dx) * 32 + Kb];
                b1.h[0] = bp[0]; b1.h[1] = bp[2];
            }
            acc = __builtin_amdgcn_wmma_f32_16x16x32_bf16(
                      false, a0.v, false, b0.v, (short)0, acc, false, false);
            a0 = a1; b0 = b1;
        }
        __syncthreads();
    }

    // ---- epilogue: bias + per-head activation, scatter to plane-major ----
    const int p    = x0 + nt * 16 + (lane & 15);
    const int cbas = (lane >= 16) ? 8 : 0;
    const size_t pix = (size_t)yy * HW_W + p;
    #pragma unroll
    for (int r = 0; r < 8; ++r) {
        const int ch = cbas + r;
        if (ch >= 11) continue;
        float bias;
        if      (ch < 3) bias = hm_b[ch];
        else if (ch < 5) bias = ce_b[ch - 3];
        else if (ch < 6) bias = cz_b[0];
        else if (ch < 9) bias = dm_b[ch - 6];
        else             bias = rt_b[ch - 9];
        float v = acc[r] + bias;
        if (ch < 3) {
            hm[((size_t)b * 3 + ch) * HW_S + pix] = 1.0f / (1.0f + expf(-v));
        } else {
            int rc = ch - 3;                        // 0,1=center 2=cz 3..5=dim 6,7=rot
            if (rc >= 3 && rc < 6) v = expf(v);     // dims: exp
            reg[((size_t)b * 8 + rc) * HW_S + pix] = v;
        }
    }
}

// ---------------------------------------------------------------------------
// Kernel 3: per-(batch,class) top-500; iterative argmax, owner-only rescan.
// ---------------------------------------------------------------------------
__global__ __launch_bounds__(256)
void topk_class_kernel(const float* __restrict__ hm, float* __restrict__ work,
                       float* __restrict__ cs, int* __restrict__ ci) {
    const int bc  = blockIdx.x;                 // b*3 + c
    const int tid = threadIdx.x;
    const size_t base = (size_t)bc * HW_S;

    for (int i = tid; i < HW_S; i += 256) work[base + i] = hm[base + i];
    __syncthreads();

    float m = -1e30f;
    int   mi = 0;
    for (int i = tid; i < HW_S; i += 256) {
        float v = work[base + i];
        if (v > m) { m = v; mi = i; }
    }

    __shared__ float lv[256];
    __shared__ int   li[256];
    __shared__ int   lt[256];

    for (int j = 0; j < TOPK; ++j) {
        lv[tid] = m; li[tid] = mi; lt[tid] = tid;
        __syncthreads();
        for (int s = 128; s > 0; s >>= 1) {
            if (tid < s && lv[tid + s] > lv[tid]) {
                lv[tid] = lv[tid + s]; li[tid] = li[tid + s]; lt[tid] = lt[tid + s];
            }
            __syncthreads();
        }
        if (tid == 0) {
            cs[bc * TOPK + j] = lv[0];
            ci[bc * TOPK + j] = li[0];
            work[base + li[0]] = -1e30f;
        }
        __syncthreads();
        if (tid == lt[0]) {             // only the owner rescans its strip
            m = -1e30f; mi = 0;
            for (int i = tid; i < HW_S; i += 256) {
                float v = work[base + i];
                if (v > m) { m = v; mi = i; }
            }
        }
        __syncthreads();
    }
}

// ---------------------------------------------------------------------------
// Kernel 4: per-batch top-500 across the 3*500 per-class candidates (in LDS).
// ---------------------------------------------------------------------------
__global__ __launch_bounds__(256)
void topk_cross_kernel(const float* __restrict__ cs, const int* __restrict__ ci,
                       float* __restrict__ ss, int* __restrict__ si,
                       int* __restrict__ scl) {
    const int b   = blockIdx.x;
    const int tid = threadIdx.x;
    __shared__ float s[3 * TOPK];
    __shared__ int   sid[3 * TOPK];
    __shared__ int   sc2[3 * TOPK];
    __shared__ float lv[256];
    __shared__ int   li[256];

    for (int e = tid; e < 3 * TOPK; e += 256) {
        int c = e / TOPK;
        int k = e % TOPK;
        s[e]   = cs[(b * 3 + c) * TOPK + k];
        sid[e] = ci[(b * 3 + c) * TOPK + k];
        sc2[e] = c;
    }
    __syncthreads();

    for (int j = 0; j < TOPK; ++j) {
        float m = -1e30f;
        int   mi = 0;
        for (int e = tid; e < 3 * TOPK; e += 256) {
            if (s[e] > m) { m = s[e]; mi = e; }
        }
        lv[tid] = m; li[tid] = mi;
        __syncthreads();
        for (int st = 128; st > 0; st >>= 1) {
            if (tid < st && lv[tid + st] > lv[tid]) {
                lv[tid] = lv[tid + st]; li[tid] = li[tid + st];
            }
            __syncthreads();
        }
        if (tid == 0) {
            int w0 = li[0];
            ss[b * TOPK + j]  = lv[0];
            si[b * TOPK + j]  = sid[w0];
            scl[b * TOPK + j] = sc2[w0];
            s[w0] = -1e30f;
        }
        __syncthreads();
    }
}

// ---------------------------------------------------------------------------
// Kernel 5: gather regressions at selected pixels, build boxes / mask, write
// the concatenated outputs (boxes | scores | class_ids | mask).
// ---------------------------------------------------------------------------
__global__ __launch_bounds__(256)
void decode_kernel(const float* __restrict__ reg,
                   const float* __restrict__ ss, const int* __restrict__ si,
                   const int* __restrict__ scl, float* __restrict__ out) {
    const int t = blockIdx.x * blockDim.x + threadIdx.x;
    if (t >= NB * TOPK) return;
    const int b = t / TOPK;

    const float score = ss[t];
    const int   idx   = si[t];
    const int   cls   = scl[t];
    const float xs = (float)(idx % HW_W);
    const float ys = (float)(idx / HW_W);

    const size_t rb = (size_t)b * 8 * HW_S + idx;
    const float c0 = reg[rb + 0 * HW_S];
    const float c1 = reg[rb + 1 * HW_S];
    const float cz = reg[rb + 2 * HW_S];
    const float d0 = reg[rb + 3 * HW_S];
    const float d1 = reg[rb + 4 * HW_S];
    const float d2 = reg[rb + 5 * HW_S];
    const float r0 = reg[rb + 6 * HW_S];
    const float r1 = reg[rb + 7 * HW_S];

    const float bx = (xs + c0) * 0.2f + 0.0f;        // STRIDE*VOXEL + PCR[0]
    const float by = (ys + c1) * 0.2f + (-44.8f);    //               PCR[1]
    const float ang = atan2f(r1, r0);

    float* bo = out + (size_t)t * 7;
    bo[0] = bx; bo[1] = by; bo[2] = cz;
    bo[3] = d0; bo[4] = d1; bo[5] = d2; bo[6] = ang;

    const float thr = (cls == 0) ? 0.2f : 0.3f;
    bool mk = (bx >= 0.0f)   && (by >= -44.8f) && (cz >= -2.0f) &&
              (bx <= 89.6f)  && (by <= 44.8f)  && (cz <= 4.0f)  &&
              (score > thr);

    out[NB * TOPK * 7 + t]                 = score;
    out[NB * TOPK * 7 + NB * TOPK + t]     = (float)cls;
    out[NB * TOPK * 7 + 2 * NB * TOPK + t] = mk ? 1.0f : 0.0f;
}

// ---------------------------------------------------------------------------
extern "C" void kernel_launch(void* const* d_in, const int* in_sizes, int n_in,
                              void* d_out, int out_size, void* d_ws, size_t ws_size,
                              hipStream_t stream) {
    const float* x     = (const float*)d_in[0];
    const float* sc_w  = (const float*)d_in[1];
    const float* sc_b  = (const float*)d_in[2];
    const float* gamma = (const float*)d_in[3];
    const float* beta  = (const float*)d_in[4];
    const float* mean  = (const float*)d_in[5];
    const float* var   = (const float*)d_in[6];
    const float* hm_w  = (const float*)d_in[7];
    const float* hm_b  = (const float*)d_in[8];
    const float* ce_w  = (const float*)d_in[9];
    const float* ce_b  = (const float*)d_in[10];
    const float* cz_w  = (const float*)d_in[11];
    const float* cz_b  = (const float*)d_in[12];
    const float* dm_w  = (const float*)d_in[13];
    const float* dm_b  = (const float*)d_in[14];
    const float* rt_w  = (const float*)d_in[15];
    const float* rt_b  = (const float*)d_in[16];
    float* out = (float*)d_out;

    // Workspace carve-up (offsets 256B-aligned)
    char* ws = (char*)d_ws;
    size_t off = 0;
    unsigned short* xp  = (unsigned short*)(ws + off); off += (size_t)NB * CIN * HW_S * 2;      // 411 MB
    unsigned short* wp  = (unsigned short*)(ws + off); off += 8 * 9 * 64 * 32 * 2;              // 288 KB
    unsigned short* wp2 = (unsigned short*)(ws + off); off += 2 * 9 * 16 * 32 * 2;              //  18 KB
    unsigned short* y_bf16 = (unsigned short*)(ws + off); off += (size_t)NB * HW_S * COUT1 * 2; // 103 MB
    float* hm   = (float*)(ws + off); off += (size_t)NB * 3 * HW_S * 4;                         // 9.6 MB
    float* reg  = (float*)(ws + off); off += (size_t)NB * 8 * HW_S * 4;                         // 26 MB
    float* work = (float*)(ws + off); off += (size_t)NB * 3 * HW_S * 4;                         // 9.6 MB
    float* cs   = (float*)(ws + off); off += 24064;   // 12 * 500 scores
    int*   ci   = (int*)  (ws + off); off += 24064;   // 12 * 500 indices
    float* ss   = (float*)(ws + off); off += 8192;    //  4 * 500 selected scores
    int*   si   = (int*)  (ws + off); off += 8192;    //  4 * 500 selected flat idx
    int*   scl  = (int*)  (ws + off); off += 8192;    //  4 * 500 selected class
    (void)ws_size; (void)in_sizes; (void)n_in; (void)out_size;

    // 0) one-time bf16 re-layouts
    pack_x_kernel<<<dim3(HW_W / 64, HW_H, NB * 8), 256, 0, stream>>>(x, xp);
    pack_w_kernel<<<dim3((8 * 9 * 64 * 32 + 255) / 256), 256, 0, stream>>>(sc_w, wp);
    pack_w2_kernel<<<dim3((2 * 9 * 16 * 32 + 255) / 256), 256, 0, stream>>>(
        hm_w, ce_w, cz_w, dm_w, rt_w, wp2);

    // 1) shared conv + BN + ReLU  (bf16 WMMA, async double-buffered LDS)
    conv1_wmma_kernel<<<dim3(HW_W / 32, HW_H, NB), 256, 0, stream>>>(
        xp, wp, sc_b, gamma, beta, mean, var, y_bf16);

    // 2) fused heads (bf16 WMMA, async LDS staging) + activations
    heads_wmma_kernel<<<dim3(HW_W / 64, HW_H, NB), 128, 0, stream>>>(
        y_bf16, wp2, hm_b, ce_b, cz_b, dm_b, rt_b, hm, reg);

    // 3) per-class top-500
    topk_class_kernel<<<dim3(NB * 3), 256, 0, stream>>>(hm, work, cs, ci);

    // 4) cross-class top-500
    topk_cross_kernel<<<dim3(NB), 256, 0, stream>>>(cs, ci, ss, si, scl);

    // 5) gather + boxes + mask -> d_out
    decode_kernel<<<dim3((NB * TOPK + 255) / 256), 256, 0, stream>>>(
        reg, ss, si, scl, out);
}
